// HNO_41102837022982
// MI455X (gfx1250) — compile-verified, matrix operand order checked by reading, not run on hardware
//
#include <hip/hip_runtime.h>

typedef __attribute__((ext_vector_type(16))) __bf16 v16bf;
typedef __attribute__((ext_vector_type(8)))  float  v8f;
typedef __attribute__((ext_vector_type(4)))  unsigned int v4u;
typedef __attribute__((ext_vector_type(8)))  int v8i;
typedef __attribute__((ext_vector_type(4)))  int v4i;

#define HIDF 128
#define NEG_SLOPE 0.01f
#define BN_EPSF 1e-5f

// ---------------- utility kernels ----------------

__global__ void k_zero(float* __restrict__ p, int n) {
    int i = blockIdx.x * blockDim.x + threadIdx.x;
    if (i < n) p[i] = 0.0f;
}

__global__ void k_initneg(float* __restrict__ d, const float* __restrict__ s, int n) {
    int i = blockIdx.x * blockDim.x + threadIdx.x;
    if (i < n) d[i] = -s[i];
}

// Convert one ChebConv weight set (4 x 128 x 128 fp32, row-major [k][n]) to
// bf16 TRANSPOSED slices ([n][k]) so the TDM can raw-copy them into LDS in
// B-fragment-friendly layout.
__global__ void k_cvtwT(const float* __restrict__ src, __bf16* __restrict__ dst, int n) {
    int i = blockIdx.x * blockDim.x + threadIdx.x;
    if (i >= n) return;
    int slice = i >> 14;            // / 16384
    int idx   = i & 16383;
    int k = idx >> 7, c = idx & 127;
    dst[slice * 16384 + c * HIDF + k] = (__bf16)src[i];
}

// ---------------- graph prep ----------------

__global__ void k_degcnt(const long long* __restrict__ src, float* __restrict__ deg, int E) {
    int e = blockIdx.x * blockDim.x + threadIdx.x;
    if (e < E) atomicAdd(&deg[(int)src[e]], 1.0f);
}

__global__ void k_dinv(const float* __restrict__ deg, float* __restrict__ dinv, int n) {
    int i = blockIdx.x * blockDim.x + threadIdx.x;
    if (i < n) {
        float d = deg[i];
        dinv[i] = (d > 0.0f) ? rsqrtf(fmaxf(d, 1.0f)) : 0.0f;
    }
}

__global__ void k_norm(const long long* __restrict__ src, const long long* __restrict__ dst,
                       const float* __restrict__ dinv, float* __restrict__ nrm, int E) {
    int e = blockIdx.x * blockDim.x + threadIdx.x;
    if (e < E) nrm[e] = -(dinv[(int)src[e]] * dinv[(int)dst[e]]);
}

// ---------------- sparse propagation (scatter-add) ----------------

// acc[dst] += scale * norm[e] * T[src], 128 features: one wave32 per edge, float4/lane
__global__ __launch_bounds__(256) void k_prop128(
    const long long* __restrict__ src, const long long* __restrict__ dst,
    const float* __restrict__ nrm, const float* __restrict__ T,
    float* __restrict__ acc, float scale, int E)
{
    int e    = blockIdx.x * 8 + (threadIdx.x >> 5);
    int lane = threadIdx.x & 31;
    if (e >= E) return;
    int s = (int)src[e], d = (int)dst[e];
    float w = scale * nrm[e];
    const float4 v = *(const float4*)(T + (size_t)s * HIDF + lane * 4);
    float* p = acc + (size_t)d * HIDF + lane * 4;
    atomicAdd(p + 0, w * v.x);
    atomicAdd(p + 1, w * v.y);
    atomicAdd(p + 2, w * v.z);
    atomicAdd(p + 3, w * v.w);
}

// 3-feature variant for the input layer
__global__ void k_prop3(const long long* __restrict__ src, const long long* __restrict__ dst,
                        const float* __restrict__ nrm, const float* __restrict__ T,
                        float* __restrict__ acc, float scale, int E)
{
    int e = blockIdx.x * blockDim.x + threadIdx.x;
    if (e >= E) return;
    int s = (int)src[e], d = (int)dst[e];
    float w = scale * nrm[e];
    atomicAdd(&acc[(size_t)d * 3 + 0], w * T[(size_t)s * 3 + 0]);
    atomicAdd(&acc[(size_t)d * 3 + 1], w * T[(size_t)s * 3 + 1]);
    atomicAdd(&acc[(size_t)d * 3 + 2], w * T[(size_t)s * 3 + 2]);
}

// ---------------- dense GEMMs ----------------

// C[n][c] += sum_f T[n][f] * W[f][c], fin = 3 (input layer; too skinny for WMMA)
__global__ void k_gemm3(const float* __restrict__ T, const float* __restrict__ W,
                        float* __restrict__ C, int nrows)
{
    int idx = blockIdx.x * blockDim.x + threadIdx.x;
    if (idx >= nrows * HIDF) return;
    int row = idx >> 7, c = idx & 127;
    const float* t = &T[(size_t)row * 3];
    C[idx] += t[0] * W[c] + t[1] * W[HIDF + c] + t[2] * W[2 * HIDF + c];
}

// C[64 x 128] += A[64 x 128] * W[128 x 128] per block, bf16 WMMA with f32 accumulate.
// W arrives pre-converted (bf16) and pre-transposed ([n][k]); it is staged into
// LDS by the Tensor Data Mover (async DMA, TENSORcnt), overlapping the VALU
// work of converting the A tile fp32->bf16.
// Block = 256 threads = 8 waves; wave w covers rowTile (w&3)*16, colBase (w>>2)*64.
__global__ __launch_bounds__(256) void k_gemm128(
    const float* __restrict__ A, const __bf16* __restrict__ WT,
    float* __restrict__ C, int nrows)
{
    __shared__ __bf16 wLds[HIDF][HIDF];   // W transposed: wLds[n][k] (32 KB)
    __shared__ __bf16 aLds[64][HIDF];     // A block, row-major       (16 KB)

    const int tid = threadIdx.x;
    const int rowBase = blockIdx.x * 64;

    // --- TDM: async copy 128x128 bf16 (32 KB) global -> LDS, issued by wave 0 ---
    if (tid == 0) {
        unsigned long long ga = (unsigned long long)WT;
        unsigned ldsOff = (unsigned)(unsigned long long)(void*)&wLds[0][0];
        v4u g0;
        g0[0] = 1u;                                   // count=1, user mode
        g0[1] = ldsOff;                               // lds_addr (bytes)
        g0[2] = (unsigned)(ga & 0xFFFFFFFFu);         // global_addr[31:0]
        g0[3] = (unsigned)((ga >> 32) & 0x1FFFFFFu)   // global_addr[56:32]
              | (2u << 30);                           // type = 2 (image)
        v8i g1;
        g1[0] = (int)(1u << 16);        // workgroup_mask=0, data_size=1 (2 bytes)
        g1[1] = (int)(128u << 16);      // tensor_dim0 = 128 (low 16)
        g1[2] = (int)(128u << 16);      // tensor_dim0 hi=0, tensor_dim1 = 128 (low 16)
        g1[3] = (int)(128u << 16);      // tensor_dim1 hi=0, tile_dim0 = 128
        g1[4] = 128;                    // tile_dim1 = 128, tile_dim2 = 0
        g1[5] = 128;                    // tensor_dim0_stride = 128 (low 32)
        g1[6] = 0;                      // stride0 hi, stride1 low = 0
        g1[7] = 0;
        v4i gz4 = {0, 0, 0, 0};
        v8i gz8 = {0, 0, 0, 0, 0, 0, 0, 0};
        __builtin_amdgcn_tensor_load_to_lds(g0, g1, gz4, gz4, gz8, 0);
        __builtin_amdgcn_s_wait_tensorcnt(0);
    }

    // --- stage A block, converting fp32 -> bf16 (pad out-of-range rows) ---
    for (int i = tid; i < 64 * HIDF; i += 256) {
        int r = i >> 7, k = i & 127;
        int gr = rowBase + r;
        float v = (gr < nrows) ? A[(size_t)gr * HIDF + k] : 0.0f;
        aLds[r][k] = (__bf16)v;
    }
    __syncthreads();

    const int wave    = tid >> 5;
    const int lane    = tid & 31;
    const int rowTile = (wave & 3) * 16;
    const int colBase = (wave >> 2) * 64;
    const int m       = lane & 15;            // M (for A) / N (for B) within tile
    const int ko      = (lane >> 4) * 8;      // K-half select per documented layout
    const int rAdd    = (lane >> 4) * 8;      // C/D row offset per documented layout

    v8f acc[4];
    #pragma unroll
    for (int t = 0; t < 4; ++t) {
        int n = colBase + t * 16 + m;
        #pragma unroll
        for (int v = 0; v < 8; ++v) {
            int gr = rowBase + rowTile + v + rAdd;
            acc[t][v] = (gr < nrows) ? C[(size_t)gr * HIDF + n] : 0.0f;
        }
    }

    #pragma unroll
    for (int ks = 0; ks < 4; ++ks) {
        const int kb = ks * 32;
        v16bf a;
        {
            const __bf16* ap = &aLds[rowTile + m][kb];
            #pragma unroll
            for (int i = 0; i < 8; ++i) { a[i] = ap[ko + i]; a[8 + i] = ap[16 + ko + i]; }
        }
        #pragma unroll
        for (int t = 0; t < 4; ++t) {
            const __bf16* bp = &wLds[colBase + t * 16 + m][kb];
            v16bf b;
            #pragma unroll
            for (int i = 0; i < 8; ++i) { b[i] = bp[ko + i]; b[8 + i] = bp[16 + ko + i]; }
            acc[t] = __builtin_amdgcn_wmma_f32_16x16x32_bf16(
                false, a, false, b, (short)0, acc[t], false, false);
        }
    }

    #pragma unroll
    for (int t = 0; t < 4; ++t) {
        int n = colBase + t * 16 + m;
        #pragma unroll
        for (int v = 0; v < 8; ++v) {
            int gr = rowBase + rowTile + v + rAdd;
            if (gr < nrows) C[(size_t)gr * HIDF + n] = acc[t][v];
        }
    }
}

// ---------------- BatchNorm (+bias, +activation) ----------------

__device__ __forceinline__ float actf(float v, int mode) {
    return (mode == 0) ? ((v >= 0.0f) ? v : NEG_SLOPE * v) : fmaxf(v, 0.0f);
}

// column sums / sums of squares of act(H + bias)
__global__ __launch_bounds__(256) void k_colstats(
    const float* __restrict__ H, const float* __restrict__ bias,
    float* __restrict__ s1, float* __restrict__ s2, int n, int mode)
{
    __shared__ float ls1[HIDF], ls2[HIDF];
    const int tid = threadIdx.x;
    if (tid < HIDF) { ls1[tid] = 0.0f; ls2[tid] = 0.0f; }
    __syncthreads();

    const int c = tid & 127, half = tid >> 7;
    const float b = bias[c];
    float sum = 0.0f, sq = 0.0f;
    for (int row = blockIdx.x * 2 + half; row < n; row += gridDim.x * 2) {
        float v = actf(H[(size_t)row * HIDF + c] + b, mode);
        sum += v; sq += v * v;
    }
    atomicAdd(&ls1[c], sum);
    atomicAdd(&ls2[c], sq);
    __syncthreads();
    if (tid < HIDF) { atomicAdd(&s1[tid], ls1[tid]); atomicAdd(&s2[tid], ls2[tid]); }
}

__global__ void k_bn_apply(const float* __restrict__ H, const float* __restrict__ bias,
                           const float* __restrict__ s1, const float* __restrict__ s2,
                           const float* __restrict__ g, const float* __restrict__ be,
                           float* __restrict__ out, int n, float invN, int mode)
{
    int idx = blockIdx.x * blockDim.x + threadIdx.x;
    if (idx >= n * HIDF) return;
    int c = idx & 127;
    float mu  = s1[c] * invN;
    float var = s2[c] * invN - mu * mu;
    float v = actf(H[idx] + bias[c], mode);
    out[idx] = (v - mu) * rsqrtf(var + BN_EPSF) * g[c] + be[c];
}

// ---------------- final: L2 normalize rows + 128x3 projection ----------------

__global__ void k_final(const float* __restrict__ H, const float* __restrict__ b4,
                        const float* __restrict__ Wrep, const float* __restrict__ brep,
                        float* __restrict__ out, int n)
{
    int row = blockIdx.x * blockDim.x + threadIdx.x;
    if (row >= n) return;
    const float* h = &H[(size_t)row * HIDF];
    float nrm = 0.0f, y0 = 0.0f, y1 = 0.0f, y2 = 0.0f;
    #pragma unroll 4
    for (int c = 0; c < HIDF; ++c) {
        float v = h[c] + b4[c];
        nrm += v * v;
        y0 += v * Wrep[c * 3 + 0];
        y1 += v * Wrep[c * 3 + 1];
        y2 += v * Wrep[c * 3 + 2];
    }
    float inv = 1.0f / fmaxf(sqrtf(nrm), 1e-12f);
    out[row * 3 + 0] = y0 * inv + brep[0];
    out[row * 3 + 1] = y1 * inv + brep[1];
    out[row * 3 + 2] = y2 * inv + brep[2];
}

// ---------------- host orchestration ----------------

extern "C" void kernel_launch(void* const* d_in, const int* in_sizes, int n_in,
                              void* d_out, int out_size, void* d_ws, size_t ws_size,
                              hipStream_t stream) {
    const float*     x    = (const float*)d_in[0];
    const long long* ei   = (const long long*)d_in[1];
    const float*     W1   = (const float*)d_in[2];
    const float*     b1   = (const float*)d_in[3];
    const float*     W2   = (const float*)d_in[4];
    const float*     b2   = (const float*)d_in[5];
    const float*     W3   = (const float*)d_in[6];
    const float*     b3   = (const float*)d_in[7];
    const float*     W4   = (const float*)d_in[8];
    const float*     b4   = (const float*)d_in[9];
    const float*     g1   = (const float*)d_in[10];
    const float*     be1  = (const float*)d_in[11];
    const float*     g2   = (const float*)d_in[12];
    const float*     be2  = (const float*)d_in[13];
    const float*     g3   = (const float*)d_in[14];
    const float*     be3  = (const float*)d_in[15];
    const float*     Wrep = (const float*)d_in[16];
    const float*     brep = (const float*)d_in[17];

    const int N = in_sizes[0] / 3;
    const int E = in_sizes[1] / 2;
    const long long* src = ei;
    const long long* dst = ei + E;

    // workspace layout
    float* ws = (float*)d_ws;
    const size_t NB = (size_t)N * HIDF;
    float* B0   = ws;
    float* B1   = B0 + NB;
    float* B2   = B1 + NB;
    float* B3   = B2 + NB;
    float* deg  = B3 + NB;
    float* dinv = deg + N;
    float* nrm  = dinv + N;
    float* t3a  = nrm + E;
    float* t3b  = t3a + (size_t)3 * N;
    float* t3c  = t3b + (size_t)3 * N;
    float* s1   = t3c + (size_t)3 * N;
    float* s2   = s1 + HIDF;
    __bf16* wbf2 = (__bf16*)(s2 + HIDF);          // 4*128*128 bf16 each
    __bf16* wbf3 = wbf2 + 4 * 16384;
    __bf16* wbf4 = wbf3 + 4 * 16384;

    const int NH = N * HIDF;
    const float invN = 1.0f / (float)N;
    auto cdiv = [](long a, long b) { return (int)((a + b - 1) / b); };

    // ---- pre-convert ChebConv weights: fp32 [k][n] -> bf16 [n][k] per slice ----
    k_cvtwT<<<cdiv(4 * 16384, 256), 256, 0, stream>>>(W2, wbf2, 4 * 16384);
    k_cvtwT<<<cdiv(4 * 16384, 256), 256, 0, stream>>>(W3, wbf3, 4 * 16384);
    k_cvtwT<<<cdiv(4 * 16384, 256), 256, 0, stream>>>(W4, wbf4, 4 * 16384);

    // ---- graph preprocessing ----
    k_zero<<<cdiv(N, 256), 256, 0, stream>>>(deg, N);
    k_degcnt<<<cdiv(E, 256), 256, 0, stream>>>(src, deg, E);
    k_dinv<<<cdiv(N, 256), 256, 0, stream>>>(deg, dinv, N);
    k_norm<<<cdiv(E, 256), 256, 0, stream>>>(src, dst, dinv, nrm, E);

    // ---- layer 1 (fin = 3): Cheb into B0 ----
    k_zero<<<cdiv(NH, 256), 256, 0, stream>>>(B0, NH);
    k_gemm3<<<cdiv(NH, 256), 256, 0, stream>>>(x, W1 + 0 * 384, B0, N);
    k_zero<<<cdiv(3 * N, 256), 256, 0, stream>>>(t3a, 3 * N);
    k_prop3<<<cdiv(E, 256), 256, 0, stream>>>(src, dst, nrm, x, t3a, 1.0f, E);
    k_gemm3<<<cdiv(NH, 256), 256, 0, stream>>>(t3a, W1 + 1 * 384, B0, N);
    k_initneg<<<cdiv(3 * N, 256), 256, 0, stream>>>(t3b, x, 3 * N);
    k_prop3<<<cdiv(E, 256), 256, 0, stream>>>(src, dst, nrm, t3a, t3b, 2.0f, E);
    k_gemm3<<<cdiv(NH, 256), 256, 0, stream>>>(t3b, W1 + 2 * 384, B0, N);
    k_initneg<<<cdiv(3 * N, 256), 256, 0, stream>>>(t3c, t3a, 3 * N);
    k_prop3<<<cdiv(E, 256), 256, 0, stream>>>(src, dst, nrm, t3b, t3c, 2.0f, E);
    k_gemm3<<<cdiv(NH, 256), 256, 0, stream>>>(t3c, W1 + 3 * 384, B0, N);
    // BN1 (leaky) -> h in B1
    k_zero<<<1, 256, 0, stream>>>(s1, 2 * HIDF);
    k_colstats<<<512, 256, 0, stream>>>(B0, b1, s1, s2, N, 0);
    k_bn_apply<<<cdiv(NH, 256), 256, 0, stream>>>(B0, b1, s1, s2, g1, be1, B1, N, invN, 0);

    // ---- layers 2..4 (fin = 128): h in B1, scratch B2/B3, out B0 ----
    const int gGemm = cdiv(N, 64);
    const int gProp = cdiv(E, 8);
    auto cheb128 = [&](const __bf16* WT) {
        k_zero<<<cdiv(NH, 256), 256, 0, stream>>>(B0, NH);
        k_gemm128<<<gGemm, 256, 0, stream>>>(B1, WT + 0 * 16384, B0, N);           // T0 = h
        k_zero<<<cdiv(NH, 256), 256, 0, stream>>>(B2, NH);
        k_prop128<<<gProp, 256, 0, stream>>>(src, dst, nrm, B1, B2, 1.0f, E);      // T1
        k_gemm128<<<gGemm, 256, 0, stream>>>(B2, WT + 1 * 16384, B0, N);
        k_initneg<<<cdiv(NH, 256), 256, 0, stream>>>(B3, B1, NH);                  // T2 = 2*L*T1 - T0
        k_prop128<<<gProp, 256, 0, stream>>>(src, dst, nrm, B2, B3, 2.0f, E);
        k_gemm128<<<gGemm, 256, 0, stream>>>(B3, WT + 2 * 16384, B0, N);
        k_initneg<<<cdiv(NH, 256), 256, 0, stream>>>(B1, B2, NH);                  // T3 = 2*L*T2 - T1
        k_prop128<<<gProp, 256, 0, stream>>>(src, dst, nrm, B3, B1, 2.0f, E);
        k_gemm128<<<gGemm, 256, 0, stream>>>(B1, WT + 3 * 16384, B0, N);
    };

    cheb128(wbf2);
    k_zero<<<1, 256, 0, stream>>>(s1, 2 * HIDF);
    k_colstats<<<512, 256, 0, stream>>>(B0, b2, s1, s2, N, 0);
    k_bn_apply<<<cdiv(NH, 256), 256, 0, stream>>>(B0, b2, s1, s2, g2, be2, B1, N, invN, 0);

    cheb128(wbf3);
    k_zero<<<1, 256, 0, stream>>>(s1, 2 * HIDF);
    k_colstats<<<512, 256, 0, stream>>>(B0, b3, s1, s2, N, 1);                      // ReLU
    k_bn_apply<<<cdiv(NH, 256), 256, 0, stream>>>(B0, b3, s1, s2, g3, be3, B1, N, invN, 1);

    cheb128(wbf4);
    k_final<<<cdiv(N, 256), 256, 0, stream>>>(B0, b4, Wrep, brep, (float*)d_out, N);
}
